// GraphTransformer_44281112822535
// MI455X (gfx1250) — matrix operand order repaired and checked by reading, not compiled, not run on hardware
//
#include <hip/hip_runtime.h>
#include <hip/hip_bf16.h>
#include <math.h>

// ---------------------------------------------------------------------------
// Graph TransformerConv x2 for MI455X (gfx1250).
//   - Dense projections (q/k/v/skip/edge per layer) in bf16 via
//     v_wmma_f32_16x16x32_bf16 (f32 accumulate). Each wave computes a 16x64
//     strip (4 N-tiles) so the A fragment is loaded once per k-step and fed
//     to 4 consecutive WMMAs.
//   - Edge phase is the roofline bottleneck (~2.5 GB gather/scatter per call
//     vs ~75 GFLOP dense): q/k/v/e stored bf16 to halve random-gather bytes,
//     and all edge-kernel loads are 16B v8bf vectors (b128) to maximize
//     request efficiency. Scatter uses f32 global atomics.
//   - Segment softmax: 3-pass (atomic float max via int/uint ordering trick,
//     exp + atomic denom, normalize + scatter-add).
// ---------------------------------------------------------------------------

typedef __bf16 bf16_t;
typedef __attribute__((ext_vector_type(16))) __bf16 v16bf;
typedef __attribute__((ext_vector_type(8)))  __bf16 v8bf;
typedef __attribute__((ext_vector_type(8)))  float  v8f;

#define NN   50000
#define NE   400000
#define FIN  22
#define EDIM 22
#define KP1  32      // padded K for FIN/EDIM
#define D1   256     // H1*C1
#define D2   512     // H2*C2
#define HH   8
#define CC1  32
#define CC2  64

// ---------------------------------------------------------------------------
__device__ __forceinline__ void atomicMaxF(float* addr, float val) {
  if (val >= 0.0f) {
    atomicMax((int*)addr, __float_as_int(val));
  } else {
    atomicMin((unsigned int*)addr, __float_as_uint(val));
  }
}

// ---------------------------------------------------------------------------
// Prep: pad rows [M,K] f32 -> [M,KPAD] bf16 (zero padded)
// ---------------------------------------------------------------------------
__global__ void pad_rows_bf16(const float* __restrict__ A, bf16_t* __restrict__ B,
                              int M, int K, int KPAD) {
  long long t = (long long)blockIdx.x * blockDim.x + threadIdx.x;
  long long total = (long long)M * KPAD;
  if (t >= total) return;
  int m  = (int)(t / KPAD);
  int kk = (int)(t - (long long)m * KPAD);
  float v = (kk < K) ? A[(size_t)m * K + kk] : 0.0f;
  B[t] = (bf16_t)v;
}

// Prep: W [K,Nc] f32 -> WT [Nc,KPAD] bf16 (transposed, zero padded)
__global__ void transpose_w_bf16(const float* __restrict__ W, bf16_t* __restrict__ WT,
                                 int K, int Nc, int KPAD) {
  long long t = (long long)blockIdx.x * blockDim.x + threadIdx.x;
  long long total = (long long)Nc * KPAD;
  if (t >= total) return;
  int n  = (int)(t / KPAD);
  int kk = (int)(t - (long long)n * KPAD);
  float v = (kk < K) ? W[(size_t)kk * Nc + n] : 0.0f;
  WT[t] = (bf16_t)v;
}

// ---------------------------------------------------------------------------
// WMMA GEMM: C[M,Nc] = A[M,KPAD](bf16) @ WT[Nc,KPAD](bf16)^T + bias
// One wave per 16x64 output strip (4 consecutive 16x16 tiles along N) so the
// A fragment is reused by 4 WMMAs per k-step. 8 waves / 256-thread block.
// A-fragment lane layout (16x32 bf16): lane L holds row (L&15),
//   K = (L>=16?16:0)+{0..15}  -> two 16B loads.
// B-fragment: lane L holds col (L&15), K = (L>=16?16:0)+{0..15} -> 32B load.
// C/D f32: acc[r] -> row tm*16 + r + (L>=16?8:0), col base + (L&15).
// ---------------------------------------------------------------------------
template <int KPAD, bool OUT_BF16>
__global__ void gemm_wmma_bf16(const bf16_t* __restrict__ A,
                               const bf16_t* __restrict__ WT,
                               const float*  __restrict__ bias,
                               void* __restrict__ Cout,
                               int M, int Nc) {
  const int lane  = threadIdx.x & 31;
  const int wave  = threadIdx.x >> 5;
  const int stripsN = Nc >> 6;                 // 64-wide strips
  const int tilesM  = M >> 4;
  const int strip = blockIdx.x * 8 + wave;     // wave-uniform: safe early-out
  if (strip >= tilesM * stripsN) return;
  const int tm = strip / stripsN;
  const int sn = strip - tm * stripsN;

  const int l15 = lane & 15;
  const int hi  = lane >> 4;                   // 0 or 1
  const int row = (tm << 4) + l15;             // A row this lane feeds
  const int n0  = (sn << 6) + l15;             // first B/C column this lane feeds

  const bf16_t* arow = A + (size_t)row * KPAD;
  const bf16_t* brow0 = WT + (size_t)(n0)      * KPAD;
  const bf16_t* brow1 = WT + (size_t)(n0 + 16) * KPAD;
  const bf16_t* brow2 = WT + (size_t)(n0 + 32) * KPAD;
  const bf16_t* brow3 = WT + (size_t)(n0 + 48) * KPAD;
  const int kbA = hi * 16;
  const int kbB = hi * 16;

  v8f acc0 = {}, acc1 = {}, acc2 = {}, acc3 = {};
#pragma unroll
  for (int k = 0; k < KPAD; k += 32) {
    if (k + 32 < KPAD) {                       // gfx1250 global_prefetch path
      __builtin_prefetch(arow + k + 32, 0, 3);
      __builtin_prefetch(brow0 + k + 32, 0, 3);
      __builtin_prefetch(brow1 + k + 32, 0, 3);
      __builtin_prefetch(brow2 + k + 32, 0, 3);
      __builtin_prefetch(brow3 + k + 32, 0, 3);
    }
    v8bf a0 = *(const v8bf*)(arow + k + kbA);
    v8bf a1 = *(const v8bf*)(arow + k + 8 + kbA);
    v16bf av;
#pragma unroll
    for (int j = 0; j < 8; ++j) { av[j] = a0[j]; av[j + 8] = a1[j]; }
    v16bf bv0 = *(const v16bf*)(brow0 + k + kbB);
    v16bf bv1 = *(const v16bf*)(brow1 + k + kbB);
    v16bf bv2 = *(const v16bf*)(brow2 + k + kbB);
    v16bf bv3 = *(const v16bf*)(brow3 + k + kbB);
    acc0 = __builtin_amdgcn_wmma_f32_16x16x32_bf16(false, av, false, bv0,
                                                   (short)0, acc0, false, false);
    acc1 = __builtin_amdgcn_wmma_f32_16x16x32_bf16(false, av, false, bv1,
                                                   (short)0, acc1, false, false);
    acc2 = __builtin_amdgcn_wmma_f32_16x16x32_bf16(false, av, false, bv2,
                                                   (short)0, acc2, false, false);
    acc3 = __builtin_amdgcn_wmma_f32_16x16x32_bf16(false, av, false, bv3,
                                                   (short)0, acc3, false, false);
  }

  const int mBase = (tm << 4) + hi * 8;
  v8f* accs[4] = {&acc0, &acc1, &acc2, &acc3};
#pragma unroll
  for (int s = 0; s < 4; ++s) {
    const int colN = n0 + s * 16;
    const float bsum = bias ? bias[colN] : 0.0f;
    if (OUT_BF16) {
      bf16_t* C = (bf16_t*)Cout;
#pragma unroll
      for (int r = 0; r < 8; ++r)
        C[(size_t)(mBase + r) * Nc + colN] = (bf16_t)((*accs[s])[r] + bsum);
    } else {
      float* C = (float*)Cout;
#pragma unroll
      for (int r = 0; r < 8; ++r)
        C[(size_t)(mBase + r) * Nc + colN] = (*accs[s])[r] + bsum;
    }
  }
}

// ---------------------------------------------------------------------------
// Segment-softmax pass 0: init per-(node,head) max/denom state
// ---------------------------------------------------------------------------
__global__ void init_softmax_state(float* __restrict__ nmax,
                                   float* __restrict__ denom, int NH) {
  int t = blockIdx.x * blockDim.x + threadIdx.x;
  if (t >= NH) return;
  nmax[t]  = -__builtin_inff();
  denom[t] = 0.0f;
}

// Pass 1: alpha[e,h] = (q[dst]·(k[src]+e))/sqrt(C); atomic max into nmax.
// All loads are 16B v8bf vectors (C multiple of 8, row offsets 16B aligned).
template <int C>
__global__ void edge_alpha_max(const bf16_t* __restrict__ q,
                               const bf16_t* __restrict__ k,
                               const bf16_t* __restrict__ eproj,
                               const int* __restrict__ src,
                               const int* __restrict__ dst,
                               float* __restrict__ alpha,
                               float* __restrict__ nmax, int EH) {
  int t = blockIdx.x * blockDim.x + threadIdx.x;
  if (t >= EH) return;
  const int D = HH * C;
  int e = t / HH, h = t - e * HH;
  int s = src[e], d = dst[e];
  const v8bf* qp = (const v8bf*)(q     + (size_t)d * D + h * C);
  const v8bf* kp = (const v8bf*)(k     + (size_t)s * D + h * C);
  const v8bf* ep = (const v8bf*)(eproj + (size_t)e * D + h * C);
  float acc = 0.0f;
#pragma unroll
  for (int g = 0; g < C / 8; ++g) {
    v8bf qv = qp[g], kv = kp[g], ev = ep[g];
#pragma unroll
    for (int j = 0; j < 8; ++j)
      acc += (float)qv[j] * ((float)kv[j] + (float)ev[j]);
  }
  acc *= rsqrtf((float)C);
  alpha[t] = acc;
  atomicMaxF(nmax + (size_t)d * HH + h, acc);
}

// Pass 2: a = exp(alpha - max[dst]); atomic add into denom; alpha <- a
__global__ void edge_exp_denom(const int* __restrict__ dst,
                               float* __restrict__ alpha,
                               const float* __restrict__ nmax,
                               float* __restrict__ denom, int EH) {
  int t = blockIdx.x * blockDim.x + threadIdx.x;
  if (t >= EH) return;
  int e = t / HH, h = t - e * HH;
  int d = dst[e];
  float a = __expf(alpha[t] - nmax[(size_t)d * HH + h]);
  alpha[t] = a;
  atomicAdd(denom + (size_t)d * HH + h, a);
}

// Pass 3: xout[dst] += (v[src]+e) * a/denom.
// Thread per (edge, 8-channel group): two b128 loads + 8 f32 atomics.
template <int C>
__global__ void edge_message(const bf16_t* __restrict__ v,
                             const bf16_t* __restrict__ eproj,
                             const int* __restrict__ src,
                             const int* __restrict__ dst,
                             const float* __restrict__ alpha,
                             const float* __restrict__ denom,
                             float* __restrict__ xout, long long total) {
  long long t = (long long)blockIdx.x * blockDim.x + threadIdx.x;
  if (t >= total) return;                      // total = E * D / 8
  const int D = HH * C;
  const int G = D / 8;                         // 8-channel groups per edge
  int e  = (int)(t / G);
  int g  = (int)(t - (long long)e * G);
  int dc = g * 8;
  int h  = dc / C;
  int s = src[e], dn = dst[e];
  float coef = alpha[(size_t)e * HH + h] /
               (denom[(size_t)dn * HH + h] + 1e-16f);
  v8bf vv = *(const v8bf*)(v     + (size_t)s * D + dc);
  v8bf ev = *(const v8bf*)(eproj + (size_t)e * D + dc);
  float* xp = xout + (size_t)dn * D + dc;
#pragma unroll
  for (int j = 0; j < 8; ++j)
    atomicAdd(xp + j, ((float)vv[j] + (float)ev[j]) * coef);
}

// relu in place (f32), optional bf16 mirror for next-layer WMMA input
__global__ void relu_cvt(float* __restrict__ x, bf16_t* __restrict__ xb,
                         long long n) {
  long long t = (long long)blockIdx.x * blockDim.x + threadIdx.x;
  if (t >= n) return;
  float v = x[t];
  v = v > 0.0f ? v : 0.0f;
  x[t] = v;
  if (xb) xb[t] = (bf16_t)v;
}

// out[n] = sigmoid(concat(x1,x2)[n] . w_out + b_out)
__global__ void final_head(const float* __restrict__ x1,
                           const float* __restrict__ x2,
                           const float* __restrict__ w,
                           const float* __restrict__ b,
                           float* __restrict__ out, int n_nodes) {
  int n = blockIdx.x * blockDim.x + threadIdx.x;
  if (n >= n_nodes) return;
  float acc = b[0];
  const float* r1 = x1 + (size_t)n * D1;
#pragma unroll 8
  for (int i = 0; i < D1; ++i) acc += r1[i] * w[i];
  const float* r2 = x2 + (size_t)n * D2;
#pragma unroll 8
  for (int i = 0; i < D2; ++i) acc += r2[i] * w[D1 + i];
  out[n] = 1.0f / (1.0f + __expf(-acc));
}

// ---------------------------------------------------------------------------
static inline unsigned cdiv(long long a, long long b) {
  return (unsigned)((a + b - 1) / b);
}

extern "C" void kernel_launch(void* const* d_in, const int* in_sizes, int n_in,
                              void* d_out, int out_size, void* d_ws, size_t ws_size,
                              hipStream_t stream) {
  (void)in_sizes; (void)n_in; (void)out_size; (void)ws_size;
  const float* x      = (const float*)d_in[0];
  const int*   eidx   = (const int*)  d_in[1];   // [2,E]: row0 src, row1 dst
  const float* eattr  = (const float*)d_in[2];
  const float* w1_q = (const float*)d_in[3],  *b1_q = (const float*)d_in[4];
  const float* w1_k = (const float*)d_in[5],  *b1_k = (const float*)d_in[6];
  const float* w1_v = (const float*)d_in[7],  *b1_v = (const float*)d_in[8];
  const float* w1_e = (const float*)d_in[9];
  const float* w1_s = (const float*)d_in[10], *b1_s = (const float*)d_in[11];
  const float* w2_q = (const float*)d_in[12], *b2_q = (const float*)d_in[13];
  const float* w2_k = (const float*)d_in[14], *b2_k = (const float*)d_in[15];
  const float* w2_v = (const float*)d_in[16], *b2_v = (const float*)d_in[17];
  const float* w2_e = (const float*)d_in[18];
  const float* w2_s = (const float*)d_in[19], *b2_s = (const float*)d_in[20];
  const float* w_o  = (const float*)d_in[21], *b_o  = (const float*)d_in[22];
  float* out = (float*)d_out;

  const int* src = eidx;
  const int* dst = eidx + NE;

  // ---- bump allocator over d_ws (total ~1.07 GB) -------------------------
  char* ws = (char*)d_ws;
  size_t off = 0;
  auto alloc = [&](size_t bytes) -> void* {
    void* p = ws + off;
    off = (off + bytes + 255) & ~(size_t)255;
    return p;
  };
  bf16_t* xb   = (bf16_t*)alloc((size_t)NN * KP1 * 2);
  bf16_t* eb   = (bf16_t*)alloc((size_t)NE * KP1 * 2);
  bf16_t* W1qT = (bf16_t*)alloc((size_t)D1 * KP1 * 2);
  bf16_t* W1kT = (bf16_t*)alloc((size_t)D1 * KP1 * 2);
  bf16_t* W1vT = (bf16_t*)alloc((size_t)D1 * KP1 * 2);
  bf16_t* W1sT = (bf16_t*)alloc((size_t)D1 * KP1 * 2);
  bf16_t* W1eT = (bf16_t*)alloc((size_t)D1 * KP1 * 2);
  bf16_t* W2qT = (bf16_t*)alloc((size_t)D2 * D1 * 2);
  bf16_t* W2kT = (bf16_t*)alloc((size_t)D2 * D1 * 2);
  bf16_t* W2vT = (bf16_t*)alloc((size_t)D2 * D1 * 2);
  bf16_t* W2sT = (bf16_t*)alloc((size_t)D2 * D1 * 2);
  bf16_t* W2eT = (bf16_t*)alloc((size_t)D2 * KP1 * 2);
  bf16_t* q1   = (bf16_t*)alloc((size_t)NN * D1 * 2);
  bf16_t* k1   = (bf16_t*)alloc((size_t)NN * D1 * 2);
  bf16_t* v1   = (bf16_t*)alloc((size_t)NN * D1 * 2);
  bf16_t* e1   = (bf16_t*)alloc((size_t)NE * D1 * 2);
  float*  x1   = (float*) alloc((size_t)NN * D1 * 4);
  bf16_t* x1b  = (bf16_t*)alloc((size_t)NN * D1 * 2);
  bf16_t* q2   = (bf16_t*)alloc((size_t)NN * D2 * 2);
  bf16_t* k2   = (bf16_t*)alloc((size_t)NN * D2 * 2);
  bf16_t* v2   = (bf16_t*)alloc((size_t)NN * D2 * 2);
  bf16_t* e2   = (bf16_t*)alloc((size_t)NE * D2 * 2);
  float*  x2   = (float*) alloc((size_t)NN * D2 * 4);
  float*  alp  = (float*) alloc((size_t)NE * HH * 4);
  float*  nmx  = (float*) alloc((size_t)NN * HH * 4);
  float*  dnm  = (float*) alloc((size_t)NN * HH * 4);

  const int B = 256;

  // ---- prep: pad/convert activations + transpose weights to bf16 ---------
  pad_rows_bf16<<<cdiv((long long)NN * KP1, B), B, 0, stream>>>(x, xb, NN, FIN, KP1);
  pad_rows_bf16<<<cdiv((long long)NE * KP1, B), B, 0, stream>>>(eattr, eb, NE, EDIM, KP1);
  transpose_w_bf16<<<cdiv((long long)D1 * KP1, B), B, 0, stream>>>(w1_q, W1qT, FIN, D1, KP1);
  transpose_w_bf16<<<cdiv((long long)D1 * KP1, B), B, 0, stream>>>(w1_k, W1kT, FIN, D1, KP1);
  transpose_w_bf16<<<cdiv((long long)D1 * KP1, B), B, 0, stream>>>(w1_v, W1vT, FIN, D1, KP1);
  transpose_w_bf16<<<cdiv((long long)D1 * KP1, B), B, 0, stream>>>(w1_s, W1sT, FIN, D1, KP1);
  transpose_w_bf16<<<cdiv((long long)D1 * KP1, B), B, 0, stream>>>(w1_e, W1eT, EDIM, D1, KP1);
  transpose_w_bf16<<<cdiv((long long)D2 * D1,  B), B, 0, stream>>>(w2_q, W2qT, D1, D2, D1);
  transpose_w_bf16<<<cdiv((long long)D2 * D1,  B), B, 0, stream>>>(w2_k, W2kT, D1, D2, D1);
  transpose_w_bf16<<<cdiv((long long)D2 * D1,  B), B, 0, stream>>>(w2_v, W2vT, D1, D2, D1);
  transpose_w_bf16<<<cdiv((long long)D2 * D1,  B), B, 0, stream>>>(w2_s, W2sT, D1, D2, D1);
  transpose_w_bf16<<<cdiv((long long)D2 * KP1, B), B, 0, stream>>>(w2_e, W2eT, EDIM, D2, KP1);

  // ---- layer 1 dense projections (WMMA) ----------------------------------
  {
    unsigned gN = cdiv((long long)(NN / 16) * (D1 / 64), 8);
    unsigned gE = cdiv((long long)(NE / 16) * (D1 / 64), 8);
    gemm_wmma_bf16<KP1, true ><<<gN, B, 0, stream>>>(xb, W1qT, b1_q, q1, NN, D1);
    gemm_wmma_bf16<KP1, true ><<<gN, B, 0, stream>>>(xb, W1kT, b1_k, k1, NN, D1);
    gemm_wmma_bf16<KP1, true ><<<gN, B, 0, stream>>>(xb, W1vT, b1_v, v1, NN, D1);
    gemm_wmma_bf16<KP1, false><<<gN, B, 0, stream>>>(xb, W1sT, b1_s, x1, NN, D1); // skip
    gemm_wmma_bf16<KP1, true ><<<gE, B, 0, stream>>>(eb, W1eT, nullptr, e1, NE, D1);
  }

  // ---- layer 1 segment softmax + aggregation -----------------------------
  {
    const int EH = NE * HH;
    init_softmax_state<<<cdiv(NN * HH, B), B, 0, stream>>>(nmx, dnm, NN * HH);
    edge_alpha_max<CC1><<<cdiv(EH, B), B, 0, stream>>>(q1, k1, e1, src, dst, alp, nmx, EH);
    edge_exp_denom<<<cdiv(EH, B), B, 0, stream>>>(dst, alp, nmx, dnm, EH);
    long long tot = (long long)NE * (D1 / 8);
    edge_message<CC1><<<cdiv(tot, B), B, 0, stream>>>(v1, e1, src, dst, alp, dnm, x1, tot);
    relu_cvt<<<cdiv((long long)NN * D1, B), B, 0, stream>>>(x1, x1b, (long long)NN * D1);
  }

  // ---- layer 2 dense projections (WMMA, K=256) ---------------------------
  {
    unsigned gN = cdiv((long long)(NN / 16) * (D2 / 64), 8);
    unsigned gE = cdiv((long long)(NE / 16) * (D2 / 64), 8);
    gemm_wmma_bf16<D1,  true ><<<gN, B, 0, stream>>>(x1b, W2qT, b2_q, q2, NN, D2);
    gemm_wmma_bf16<D1,  true ><<<gN, B, 0, stream>>>(x1b, W2kT, b2_k, k2, NN, D2);
    gemm_wmma_bf16<D1,  true ><<<gN, B, 0, stream>>>(x1b, W2vT, b2_v, v2, NN, D2);
    gemm_wmma_bf16<D1,  false><<<gN, B, 0, stream>>>(x1b, W2sT, b2_s, x2, NN, D2); // skip
    gemm_wmma_bf16<KP1, true ><<<gE, B, 0, stream>>>(eb, W2eT, nullptr, e2, NE, D2);
  }

  // ---- layer 2 segment softmax + aggregation -----------------------------
  {
    const int EH = NE * HH;
    init_softmax_state<<<cdiv(NN * HH, B), B, 0, stream>>>(nmx, dnm, NN * HH);
    edge_alpha_max<CC2><<<cdiv(EH, B), B, 0, stream>>>(q2, k2, e2, src, dst, alp, nmx, EH);
    edge_exp_denom<<<cdiv(EH, B), B, 0, stream>>>(dst, alp, nmx, dnm, EH);
    long long tot = (long long)NE * (D2 / 8);
    edge_message<CC2><<<cdiv(tot, B), B, 0, stream>>>(v2, e2, src, dst, alp, dnm, x2, tot);
    relu_cvt<<<cdiv((long long)NN * D2, B), B, 0, stream>>>(x2, nullptr, (long long)NN * D2);
  }

  // ---- output head: sigmoid(concat(x1,x2) @ w_out + b_out) ---------------
  final_head<<<cdiv(NN, B), B, 0, stream>>>(x1, x2, w_o, b_o, out, NN);
}
